// ContrastiveLoss_85521388798178
// MI455X (gfx1250) — matrix operand order repaired and checked by reading, not compiled, not run on hardware
//
#include <hip/hip_runtime.h>

#define BN 8192
#define CN 16384
#define D  512

typedef __attribute__((ext_vector_type(16))) __bf16 bf16x16;
typedef __attribute__((ext_vector_type(8)))  float  f32x8;

// ---------- helpers ----------
__device__ inline unsigned short f2bf(float f) {
    unsigned int u = __float_as_uint(f);
    unsigned int r = u + 0x7FFFu + ((u >> 16) & 1u);   // round-to-nearest-even
    return (unsigned short)(r >> 16);
}

__device__ inline float waveReduce32(float v) {
    #pragma unroll
    for (int s = 16; s > 0; s >>= 1) v += __shfl_xor(v, s, 32);
    return v;
}

// 128-thread block reduce (4 waves). 'red' must be a distinct float[4] per call site.
__device__ inline float blockReduce128(float v, float* red, int tid) {
    v = waveReduce32(v);
    if ((tid & 31) == 0) red[tid >> 5] = v;
    __syncthreads();
    return red[0] + red[1] + red[2] + red[3];
}

// ---------- kernel 1: normalize negatives -> bf16 ----------
__global__ __launch_bounds__(128)
void norm_neg_kernel(const float* __restrict__ neg, unsigned short* __restrict__ nn) {
    const int row = blockIdx.x;
    const int tid = threadIdx.x;
    const float4 v = *(const float4*)(neg + (size_t)row * D + tid * 4);
    __shared__ float red[4];
    float ss = blockReduce128(v.x*v.x + v.y*v.y + v.z*v.z + v.w*v.w, red, tid);
    const float inv = 1.0f / fmaxf(sqrtf(ss), 1e-12f);
    ushort4 o = make_ushort4(f2bf(v.x*inv), f2bf(v.y*inv), f2bf(v.z*inv), f2bf(v.w*inv));
    *(ushort4*)(nn + (size_t)row * D + tid * 4) = o;
}

// ---------- kernel 2: normalize x -> bf16, sim = <xn, pn>, denom = 0 ----------
__global__ __launch_bounds__(128)
void norm_x_kernel(const float* __restrict__ x, const float* __restrict__ pos,
                   unsigned short* __restrict__ xn, float* __restrict__ sim,
                   float* __restrict__ denom) {
    const int row = blockIdx.x;
    const int tid = threadIdx.x;
    const float4 xv = *(const float4*)(x   + (size_t)row * D + tid * 4);
    const float4 pv = *(const float4*)(pos + (size_t)row * D + tid * 4);
    __shared__ float r0[4], r1[4], r2[4];
    float ssx = blockReduce128(xv.x*xv.x + xv.y*xv.y + xv.z*xv.z + xv.w*xv.w, r0, tid);
    float ssp = blockReduce128(pv.x*pv.x + pv.y*pv.y + pv.z*pv.z + pv.w*pv.w, r1, tid);
    float dxp = blockReduce128(xv.x*pv.x + xv.y*pv.y + xv.z*pv.z + xv.w*pv.w, r2, tid);
    const float invx = 1.0f / fmaxf(sqrtf(ssx), 1e-12f);
    const float invp = 1.0f / fmaxf(sqrtf(ssp), 1e-12f);
    ushort4 o = make_ushort4(f2bf(xv.x*invx), f2bf(xv.y*invx), f2bf(xv.z*invx), f2bf(xv.w*invx));
    *(ushort4*)(xn + (size_t)row * D + tid * 4) = o;
    if (tid == 0) {
        sim[row]   = dxp * invx * invp;
        denom[row] = 0.0f;                 // re-init every launch (graph replay safe)
    }
}

// ---------- kernel 3: WMMA GEMM (xn @ nn^T) fused with exp-rowsum ----------
// block = 256 threads = 8 waves. Block tile 128(M) x 256(N).
// Wave grid 2(M) x 4(N): each wave owns a 64x64 register tile = 4x4 WMMA tiles.
// Per K-step: 16 ds_load_b128 feed 16 v_wmma (ratio 1.0).
// K walked in 32-wide steps through LDS (rows padded to 40 bf16 = 80B, conflict-free).
__global__ __launch_bounds__(256)
void gemm_expsum_kernel(const unsigned short* __restrict__ xn,
                        const unsigned short* __restrict__ nn,
                        float* __restrict__ denom) {
    __shared__ __align__(16) unsigned short As[128 * 40];
    __shared__ __align__(16) unsigned short Bs[256 * 40];
    __shared__ float denomP[128];

    const int tid  = threadIdx.x;
    const int lane = tid & 31;
    const int w    = tid >> 5;        // 0..7
    const int wm   = w & 1;           // 2 M-groups of 64 rows
    const int wn   = w >> 1;          // 4 N-groups of 64 cols
    const int hp   = lane >> 4;       // half-wave: 0 or 1
    const int l16  = lane & 15;
    const int rowBase = blockIdx.x * 128;
    const int colBase = blockIdx.y * 256;

    if (tid < 128) denomP[tid] = 0.0f;

    f32x8 acc[4][4];
    #pragma unroll
    for (int m = 0; m < 4; ++m)
        #pragma unroll
        for (int n = 0; n < 4; ++n)
            #pragma unroll
            for (int g = 0; g < 8; ++g) acc[m][n][g] = 0.0f;

    for (int k0 = 0; k0 < D; k0 += 32) {
        __syncthreads();
        // cooperative G->LDS: A = 128x32 bf16, B = 256x32 bf16, 16B chunks
        #pragma unroll
        for (int i = 0; i < 2; ++i) {
            const int c  = tid + 256 * i;          // 0..511
            const int r  = c >> 2;
            const int cc = c & 3;
            *(uint4*)(As + r * 40 + cc * 8) =
                *(const uint4*)(xn + (size_t)(rowBase + r) * D + k0 + cc * 8);
        }
        #pragma unroll
        for (int i = 0; i < 4; ++i) {
            const int c  = tid + 256 * i;          // 0..1023
            const int r  = c >> 2;
            const int cc = c & 3;
            *(uint4*)(Bs + r * 40 + cc * 8) =
                *(const uint4*)(nn + (size_t)(colBase + r) * D + k0 + cc * 8);
        }
        __syncthreads();

        union { bf16x16 v; uint4 q[2]; } a[4], b[4];
        // A 16x32 bf16 layout: lanes 0-15 -> K[0..7]+K[16..23]; lanes 16-31 -> K[8..15]+K[24..31]
        #pragma unroll
        for (int m = 0; m < 4; ++m) {
            const unsigned short* base = As + (wm * 64 + m * 16 + l16) * 40;
            a[m].q[0] = *(const uint4*)(base + hp * 8);
            a[m].q[1] = *(const uint4*)(base + 16 + hp * 8);
        }
        // B 32x16 bf16 layout: lanes 0-15 -> K[0..7]+K[16..23]; lanes 16-31 -> K[8..15]+K[24..31]
        #pragma unroll
        for (int n = 0; n < 4; ++n) {
            const unsigned short* base = Bs + (wn * 64 + n * 16 + l16) * 40;
            b[n].q[0] = *(const uint4*)(base + hp * 8);
            b[n].q[1] = *(const uint4*)(base + 16 + hp * 8);
        }
        #pragma unroll
        for (int m = 0; m < 4; ++m)
            #pragma unroll
            for (int n = 0; n < 4; ++n)
                acc[m][n] = __builtin_amdgcn_wmma_f32_16x16x32_bf16(
                    false, a[m].v, false, b[n].v, (short)0, acc[m][n], false, false);
    }

    // epilogue: denom[row] += sum_N exp(logit)
    __syncthreads();
    #pragma unroll
    for (int m = 0; m < 4; ++m) {
        #pragma unroll
        for (int g = 0; g < 8; ++g) {
            float e = 0.0f;
            #pragma unroll
            for (int n = 0; n < 4; ++n) e += __expf(acc[m][n][g]);
            #pragma unroll
            for (int s = 1; s < 16; s <<= 1) e += __shfl_xor(e, s, 16);
            if (l16 == 0) {
                const int row = wm * 64 + m * 16 + g + 8 * hp;  // C layout: M = g + 8*half
                atomicAdd(&denomP[row], e);                     // LDS f32 atomic
            }
        }
    }
    __syncthreads();
    if (tid < 128) atomicAdd(&denom[rowBase + tid], denomP[tid]);
}

// ---------- kernel 4: loss = -(1/BN) * sum(sim - log(denom)) ----------
__global__ __launch_bounds__(256)
void loss_kernel(const float* __restrict__ sim, const float* __restrict__ denom,
                 float* __restrict__ out) {
    __shared__ float red[256];
    float s = 0.0f;
    for (int i = threadIdx.x; i < BN; i += 256)
        s += sim[i] - __logf(denom[i]);
    red[threadIdx.x] = s;
    __syncthreads();
    for (int st = 128; st > 0; st >>= 1) {
        if (threadIdx.x < st) red[threadIdx.x] += red[threadIdx.x + st];
        __syncthreads();
    }
    if (threadIdx.x == 0) out[0] = -red[0] / (float)BN;
}

// ---------- launch ----------
extern "C" void kernel_launch(void* const* d_in, const int* in_sizes, int n_in,
                              void* d_out, int out_size, void* d_ws, size_t ws_size,
                              hipStream_t stream) {
    const float* x   = (const float*)d_in[0];
    const float* pos = (const float*)d_in[1];
    const float* neg = (const float*)d_in[2];

    // workspace layout
    unsigned short* xn = (unsigned short*)d_ws;                       // 8192*512 bf16 =  8 MB
    unsigned short* nn = xn + (size_t)BN * D;                         // 16384*512 bf16 = 16 MB
    float* sim   = (float*)((char*)d_ws + (size_t)(BN + CN) * D * 2); // 32 KB
    float* denom = sim + BN;                                          // 32 KB
    float* out   = (float*)d_out;

    norm_neg_kernel<<<CN, 128, 0, stream>>>(neg, nn);
    norm_x_kernel<<<BN, 128, 0, stream>>>(x, pos, xn, sim, denom);

    dim3 grid(BN / 128, CN / 256);   // 64 x 64 blocks
    gemm_expsum_kernel<<<grid, 256, 0, stream>>>(xn, nn, denom);

    loss_kernel<<<1, 256, 0, stream>>>(sim, denom, out);
}
